// GoldenMoELayer_59210419142730
// MI455X (gfx1250) — matrix operand order repaired and checked by reading, not compiled, not run on hardware
//
#include <hip/hip_runtime.h>

typedef __attribute__((ext_vector_type(16))) _Float16 v16h;
typedef __attribute__((ext_vector_type(8)))  _Float16 v8h;
typedef __attribute__((ext_vector_type(8)))  float    v8f;
typedef __attribute__((ext_vector_type(4)))  int      v4i;

#define LDSTRIDE 40   // halfs per LDS row: 32 data + 8 pad (80B rows, conflict-free b128)

#define GOLDEN_CENTER 0.36787944117144233f
#define GOLDEN_LOWER  0.21231792754821910f
#define GOLDEN_UPPER  0.5f

// -------- optional CDNA5 async global->LDS path (ASYNCcnt), compile-safe fallback --------
#if defined(__has_builtin)
#  if __has_builtin(__builtin_amdgcn_global_load_async_to_lds_b128) && \
      __has_builtin(__builtin_amdgcn_s_wait_asynccnt)
#    define USE_ASYNC_LDS 1
#  endif
#endif
#ifndef USE_ASYNC_LDS
#  define USE_ASYNC_LDS 0
#endif

typedef __attribute__((address_space(1))) v4i gv4i;
typedef __attribute__((address_space(3))) v4i lv4i;

// ---------------- fragment loaders (CDNA5 WMMA 16x16x32 f16 layouts) ----------------
// A (16x32, MxK): lanes 0-15 hold M=lane, K=0..7 (v0-3) and K=16..23 (v4-7);
//                 lanes 16-31 hold M=lane-16, K=8..15 and K=24..31.
__device__ __forceinline__ v16h frag_a(const _Float16* tile, int lane) {
  const int row  = lane & 15;
  const int koff = (lane >> 4) << 3;          // 0 or 8
  const _Float16* p = tile + row * LDSTRIDE + koff;
  v16h r;
  ((v8h*)&r)[0] = *(const v8h*)(p);           // K = koff .. koff+7
  ((v8h*)&r)[1] = *(const v8h*)(p + 16);      // K = koff+16 .. koff+23
  return r;
}
// B (32x16, KxN), B[k][n] = W[n][k] with W row-major over k:
// lanes 0-15: N=lane, K=0..7 and K=16..23; lanes 16-31: N=lane-16, K=8..15 and K=24..31.
__device__ __forceinline__ v16h frag_b(const _Float16* tile, int lane) {
  const int row  = lane & 15;
  const int koff = (lane >> 4) << 3;          // 0 or 8
  const _Float16* p = tile + row * LDSTRIDE + koff;
  v16h r;
  ((v8h*)&r)[0] = *(const v8h*)(p);
  ((v8h*)&r)[1] = *(const v8h*)(p + 16);
  return r;
}

// Stage a 64x32 f16 tile (row-major, src stride in halfs) into padded LDS. 128 threads.
__device__ __forceinline__ void stage_sync(_Float16* dst, const _Float16* src,
                                           int src_stride, int tid) {
#pragma unroll
  for (int i = 0; i < 2; ++i) {
    const int c   = tid + i * 128;            // 256 chunks of 8 halfs
    const int row = c >> 2;
    const int col = (c & 3) * 8;
    *(v8h*)(dst + row * LDSTRIDE + col) = *(const v8h*)(src + (size_t)row * src_stride + col);
  }
}

#if USE_ASYNC_LDS
__device__ __forceinline__ void stage_async(_Float16* dst, const _Float16* src,
                                            int src_stride, int tid) {
#pragma unroll
  for (int i = 0; i < 2; ++i) {
    const int c   = tid + i * 128;
    const int row = c >> 2;
    const int col = (c & 3) * 8;
    __builtin_amdgcn_global_load_async_to_lds_b128(
        (gv4i*)(src + (size_t)row * src_stride + col),
        (lv4i*)(dst + row * LDSTRIDE + col),
        0, 0);
  }
}
#  define STAGE stage_async
#  define ASYNC_WAIT() __builtin_amdgcn_s_wait_asynccnt(0)
#else
#  define STAGE stage_sync
#  define ASYNC_WAIT() ((void)0)
#endif

// ---------------- prep kernels ----------------
__global__ void convert_x_kernel(const float* __restrict__ x, _Float16* __restrict__ xh) {
  const size_t i = (size_t)blockIdx.x * blockDim.x + threadIdx.x;   // 2048*1024 exact
  xh[i] = (_Float16)x[i];
}

// out[e,row,col] = W[e,row,col] + (e<2 ? sum_r la_b[e,row,r]*la_a[e,r,col] : 0), as f16.
__global__ void prep_weight_kernel(const float* __restrict__ W,
                                   const float* __restrict__ la_a,  // [2,64,ncols]
                                   const float* __restrict__ la_b,  // [2,nrows,64]
                                   _Float16* __restrict__ out,
                                   int nrows, int ncols) {
  const size_t idx   = (size_t)blockIdx.x * blockDim.x + threadIdx.x;  // 8*nrows*ncols exact
  const size_t per_e = (size_t)nrows * ncols;
  const int e   = (int)(idx / per_e);
  const size_t r2 = idx - (size_t)e * per_e;
  const int row = (int)(r2 / ncols);
  const int col = (int)(r2 % ncols);
  float v = W[idx];
  if (e < 2) {
    const float* bp = la_b + ((size_t)e * nrows + row) * 64;
    const float* ap = la_a + (size_t)e * 64 * ncols + col;
    float d = 0.f;
#pragma unroll 8
    for (int r = 0; r < 64; ++r) d += bp[r] * ap[(size_t)r * ncols];
    v += d;
  }
  out[idx] = (_Float16)v;
}

// ---------------- golden-ratio router: one wave (32 lanes) per token ----------------
__global__ void __launch_bounds__(128) router_kernel(const float* __restrict__ x,
                                                     const float* __restrict__ Wr,
                                                     const float* __restrict__ temp,
                                                     float* __restrict__ wout) {
  const int token = blockIdx.x * 4 + (threadIdx.x >> 5);
  const int lane  = threadIdx.x & 31;
  const float* xr = x + (size_t)token * 1024;
  float acc[8];
#pragma unroll
  for (int e = 0; e < 8; ++e) acc[e] = 0.f;
  for (int d = lane; d < 1024; d += 32) {
    const float xv = xr[d];
#pragma unroll
    for (int e = 0; e < 8; ++e) acc[e] += xv * Wr[e * 1024 + d];
  }
#pragma unroll
  for (int off = 16; off > 0; off >>= 1)
#pragma unroll
    for (int e = 0; e < 8; ++e) acc[e] += __shfl_xor(acc[e], off, 32);

  if (lane == 0) {
    const float t = temp[0];
    float w[8], fb[8];
    float wsum = 0.f;
#pragma unroll
    for (int e = 0; e < 8; ++e) {
      const float inh  = 1.f / (1.f + expf(-acc[e] / t));
      const float dist = fabsf(inh - GOLDEN_CENTER);
      const bool zone  = (inh >= GOLDEN_LOWER) && (inh <= GOLDEN_UPPER);
      w[e]  = zone ? expf(-dist * 10.f) : 0.f;
      fb[e] = expf(-dist * (1.f / 0.3f));
      wsum += w[e];
    }
    int i1 = 0;
#pragma unroll
    for (int e = 1; e < 8; ++e) if (fb[e] > fb[i1]) i1 = e;   // strict > == jax tie-break
    int i2 = (i1 == 0) ? 1 : 0;
#pragma unroll
    for (int e = 0; e < 8; ++e) if (e != i1 && fb[e] > fb[i2]) i2 = e;
    const float fbs  = fmaxf(fb[i1] + fb[i2], 1e-8f);
    const bool noexp = wsum < 1e-8f;
    float s2 = 0.f;
#pragma unroll
    for (int e = 0; e < 8; ++e) {
      const float fbw = (e == i1) ? fb[i1] / fbs : ((e == i2) ? fb[i2] / fbs : 0.f);
      w[e] = noexp ? fbw : w[e];
      s2  += w[e];
    }
    s2 = fmaxf(s2, 1e-8f);
#pragma unroll
    for (int e = 0; e < 8; ++e) wout[token * 8 + e] = w[e] / s2;
  }
}

// ---------------- gate/up GEMMs + SiLU, per expert (double-buffered) ----------------
// grid (EI/64, 2048/64, E), block 128 (4 waves in 2x2, each 32x32 quadrant)
__global__ void __launch_bounds__(128) gateup_kernel(const _Float16* __restrict__ xh,
                                                     const _Float16* __restrict__ wgh,
                                                     const _Float16* __restrict__ wuh,
                                                     _Float16* __restrict__ midh) {
  __shared__ __align__(16) _Float16 xs[2][64 * LDSTRIDE];
  __shared__ __align__(16) _Float16 gs[2][64 * LDSTRIDE];
  __shared__ __align__(16) _Float16 us[2][64 * LDSTRIDE];
  const int tid = threadIdx.x, lane = tid & 31, wid = tid >> 5;
  const int wm = wid & 1, wn = wid >> 1;
  const int e  = blockIdx.z;
  const int m0 = blockIdx.y * 64, n0 = blockIdx.x * 64;
  const _Float16* xg = xh  + (size_t)m0 * 1024;
  const _Float16* wg = wgh + ((size_t)e * 1024 + n0) * 1024;
  const _Float16* wu = wuh + ((size_t)e * 1024 + n0) * 1024;

  const v8f zero = {0.f, 0.f, 0.f, 0.f, 0.f, 0.f, 0.f, 0.f};
  v8f cg[2][2], cu[2][2];
#pragma unroll
  for (int i = 0; i < 2; ++i)
#pragma unroll
    for (int j = 0; j < 2; ++j) { cg[i][j] = zero; cu[i][j] = zero; }

  // prologue: stage k=0 into buffer 0
  STAGE(xs[0], xg, 1024, tid);
  STAGE(gs[0], wg, 1024, tid);
  STAGE(us[0], wu, 1024, tid);
  ASYNC_WAIT();
  __syncthreads();

  for (int k0 = 0; k0 < 1024; k0 += 32) {
    const int cur = (k0 >> 5) & 1, nxt = cur ^ 1;
    const bool more = (k0 + 32) < 1024;
    if (more) {                               // issue next stage (overlaps compute)
      STAGE(xs[nxt], xg + k0 + 32, 1024, tid);
      STAGE(gs[nxt], wg + k0 + 32, 1024, tid);
      STAGE(us[nxt], wu + k0 + 32, 1024, tid);
#if !USE_ASYNC_LDS
      if (k0 + 64 < 1024) {                   // gfx1250 global_prefetch_b8, 2 stages ahead
        const size_t po = (size_t)(tid >> 2) * 1024 + (tid & 3) * 8 + k0 + 64;
        __builtin_prefetch(xg + po, 0, 0);
        __builtin_prefetch(wg + po, 0, 0);
        __builtin_prefetch(wu + po, 0, 0);
      }
#endif
    }
    const v16h a0  = frag_a(xs[cur] + (wm * 32)      * LDSTRIDE, lane);
    const v16h a1  = frag_a(xs[cur] + (wm * 32 + 16) * LDSTRIDE, lane);
    const v16h bg0 = frag_b(gs[cur] + (wn * 32)      * LDSTRIDE, lane);
    const v16h bg1 = frag_b(gs[cur] + (wn * 32 + 16) * LDSTRIDE, lane);
    const v16h bu0 = frag_b(us[cur] + (wn * 32)      * LDSTRIDE, lane);
    const v16h bu1 = frag_b(us[cur] + (wn * 32 + 16) * LDSTRIDE, lane);
    cg[0][0] = __builtin_amdgcn_wmma_f32_16x16x32_f16(false, a0, false, bg0, (short)0, cg[0][0], false, false);
    cg[0][1] = __builtin_amdgcn_wmma_f32_16x16x32_f16(false, a0, false, bg1, (short)0, cg[0][1], false, false);
    cg[1][0] = __builtin_amdgcn_wmma_f32_16x16x32_f16(false, a1, false, bg0, (short)0, cg[1][0], false, false);
    cg[1][1] = __builtin_amdgcn_wmma_f32_16x16x32_f16(false, a1, false, bg1, (short)0, cg[1][1], false, false);
    cu[0][0] = __builtin_amdgcn_wmma_f32_16x16x32_f16(false, a0, false, bu0, (short)0, cu[0][0], false, false);
    cu[0][1] = __builtin_amdgcn_wmma_f32_16x16x32_f16(false, a0, false, bu1, (short)0, cu[0][1], false, false);
    cu[1][0] = __builtin_amdgcn_wmma_f32_16x16x32_f16(false, a1, false, bu0, (short)0, cu[1][0], false, false);
    cu[1][1] = __builtin_amdgcn_wmma_f32_16x16x32_f16(false, a1, false, bu1, (short)0, cu[1][1], false, false);
    if (more) { ASYNC_WAIT(); __syncthreads(); }
  }

  // C layout: lane l, elem v -> row = v + 8*(l>=16), col = l&15
  const int hi = lane >> 4, nl = lane & 15;
#pragma unroll
  for (int mi = 0; mi < 2; ++mi)
#pragma unroll
    for (int ni = 0; ni < 2; ++ni)
#pragma unroll
      for (int v = 0; v < 8; ++v) {
        const float g = cg[mi][ni][v], u = cu[mi][ni][v];
        const float m = g / (1.f + expf(-g)) * u;   // silu(g)*u
        const int row = m0 + wm * 32 + mi * 16 + v + hi * 8;
        const int col = n0 + wn * 32 + ni * 16 + nl;
        midh[((size_t)e * 2048 + row) * 1024 + col] = (_Float16)m;
      }
}

// ---------------- down GEMM + router-weighted combine (flattened e*k pipeline) ----------------
// grid (D/64, 2048/64), block 128
__global__ void __launch_bounds__(128) down_kernel(const _Float16* __restrict__ midh,
                                                   const _Float16* __restrict__ wdh,
                                                   const float* __restrict__ wrout,
                                                   float* __restrict__ out) {
  __shared__ __align__(16) _Float16 ms[2][64 * LDSTRIDE];
  __shared__ __align__(16) _Float16 bs[2][64 * LDSTRIDE];
  __shared__ __align__(16) float    wl[64 * 8];
  const int tid = threadIdx.x, lane = tid & 31, wid = tid >> 5;
  const int wm = wid & 1, wn = wid >> 1;
  const int m0 = blockIdx.y * 64, n0 = blockIdx.x * 64;

  *(float4*)&wl[tid * 4] = *(const float4*)&wrout[m0 * 8 + tid * 4];  // 64 tokens x 8 experts

  const v8f zero = {0.f, 0.f, 0.f, 0.f, 0.f, 0.f, 0.f, 0.f};
  v8f facc[2][2], ce[2][2];
#pragma unroll
  for (int i = 0; i < 2; ++i)
#pragma unroll
    for (int j = 0; j < 2; ++j) facc[i][j] = zero;

  const int hi = lane >> 4, nl = lane & 15;

  // prologue: stage step 0 (e=0, k=0)
  STAGE(ms[0], midh + (size_t)m0 * 1024, 1024, tid);
  STAGE(bs[0], wdh  + (size_t)n0 * 1024, 1024, tid);
  ASYNC_WAIT();
  __syncthreads();

  for (int s = 0; s < 256; ++s) {             // s = e*32 + k-step
    const int cur = s & 1, nxt = cur ^ 1;
    const int e = s >> 5, kq = s & 31;
    if (kq == 0) {
#pragma unroll
      for (int i = 0; i < 2; ++i)
#pragma unroll
        for (int j = 0; j < 2; ++j) ce[i][j] = zero;
    }
    if (s + 1 < 256) {                        // issue next stage
      const int e2 = (s + 1) >> 5, k2 = ((s + 1) & 31) * 32;
      STAGE(ms[nxt], midh + ((size_t)e2 * 2048 + m0) * 1024 + k2, 1024, tid);
      STAGE(bs[nxt], wdh  + ((size_t)e2 * 1024 + n0) * 1024 + k2, 1024, tid);
    }
    const v16h a0 = frag_a(ms[cur] + (wm * 32)      * LDSTRIDE, lane);
    const v16h a1 = frag_a(ms[cur] + (wm * 32 + 16) * LDSTRIDE, lane);
    const v16h b0 = frag_b(bs[cur] + (wn * 32)      * LDSTRIDE, lane);
    const v16h b1 = frag_b(bs[cur] + (wn * 32 + 16) * LDSTRIDE, lane);
    ce[0][0] = __builtin_amdgcn_wmma_f32_16x16x32_f16(false, a0, false, b0, (short)0, ce[0][0], false, false);
    ce[0][1] = __builtin_amdgcn_wmma_f32_16x16x32_f16(false, a0, false, b1, (short)0, ce[0][1], false, false);
    ce[1][0] = __builtin_amdgcn_wmma_f32_16x16x32_f16(false, a1, false, b0, (short)0, ce[1][0], false, false);
    ce[1][1] = __builtin_amdgcn_wmma_f32_16x16x32_f16(false, a1, false, b1, (short)0, ce[1][1], false, false);
    if (kq == 31) {                           // fold expert into weighted accumulator
#pragma unroll
      for (int mi = 0; mi < 2; ++mi)
#pragma unroll
        for (int ni = 0; ni < 2; ++ni)
#pragma unroll
          for (int v = 0; v < 8; ++v) {
            const int lrow = wm * 32 + mi * 16 + v + hi * 8;
            facc[mi][ni][v] = facc[mi][ni][v] + wl[lrow * 8 + e] * ce[mi][ni][v];
          }
    }
    if (s + 1 < 256) { ASYNC_WAIT(); __syncthreads(); }
  }

#pragma unroll
  for (int mi = 0; mi < 2; ++mi)
#pragma unroll
    for (int ni = 0; ni < 2; ++ni)
#pragma unroll
      for (int v = 0; v < 8; ++v) {
        const int row = m0 + wm * 32 + mi * 16 + v + hi * 8;
        const int col = n0 + wn * 32 + ni * 16 + nl;
        out[(size_t)row * 1024 + col] = facc[mi][ni][v];
      }
}

// ---------------- launcher ----------------
extern "C" void kernel_launch(void* const* d_in, const int* in_sizes, int n_in,
                              void* d_out, int out_size, void* d_ws, size_t ws_size,
                              hipStream_t stream) {
  (void)in_sizes; (void)n_in; (void)out_size; (void)ws_size;
  const float* x    = (const float*)d_in[0];
  const float* Wg   = (const float*)d_in[1];
  const float* Wu   = (const float*)d_in[2];
  const float* Wd   = (const float*)d_in[3];
  const float* Wr   = (const float*)d_in[4];
  const float* temp = (const float*)d_in[5];
  const float* lga  = (const float*)d_in[6];
  const float* lgb  = (const float*)d_in[7];
  const float* lua  = (const float*)d_in[8];
  const float* lub  = (const float*)d_in[9];
  const float* lda  = (const float*)d_in[10];
  const float* ldb  = (const float*)d_in[11];
  float* out = (float*)d_out;
  char*  ws  = (char*)d_ws;

  _Float16* xh   = (_Float16*)(ws);                         //  4 MB  [0, 4)
  _Float16* wgh  = (_Float16*)(ws + ((size_t)4  << 20));    // 16 MB  [4, 20)
  _Float16* wuh  = (_Float16*)(ws + ((size_t)20 << 20));    // 16 MB  [20, 36)
  _Float16* wdh  = (_Float16*)(ws + ((size_t)36 << 20));    // 16 MB  [36, 52)
  _Float16* midh = (_Float16*)(ws + ((size_t)52 << 20));    // 32 MB  [52, 84)
  float*    wr   = (float*)   (ws + ((size_t)84 << 20));    // 64 KB

  convert_x_kernel<<<8192, 256, 0, stream>>>(x, xh);
  prep_weight_kernel<<<32768, 256, 0, stream>>>(Wg, lga, lgb, wgh, 1024, 1024);
  prep_weight_kernel<<<32768, 256, 0, stream>>>(Wu, lua, lub, wuh, 1024, 1024);
  prep_weight_kernel<<<32768, 256, 0, stream>>>(Wd, lda, ldb, wdh, 1024, 1024);
  router_kernel<<<512, 128, 0, stream>>>(x, Wr, temp, wr);
  gateup_kernel<<<dim3(16, 32, 8), 128, 0, stream>>>(xh, wgh, wuh, midh);
  down_kernel<<<dim3(16, 32), 128, 0, stream>>>(midh, wdh, wr, out);
}
